// Attention_481036337444
// MI455X (gfx1250) — matrix operand order repaired and checked by reading, not compiled, not run on hardware
//
#include <hip/hip_runtime.h>
#include <hip/hip_bf16.h>

// ---------------- types ----------------
typedef __attribute__((ext_vector_type(16))) __bf16 v16bf;
typedef __attribute__((ext_vector_type(8)))  __bf16 v8bf;
typedef __attribute__((ext_vector_type(8)))  float  v8f;
typedef __attribute__((ext_vector_type(4)))  unsigned int v4u;
typedef __attribute__((ext_vector_type(8)))  int v8i;
typedef __attribute__((ext_vector_type(4)))  int v4i;

#define NB 4
#define NC 128
#define NN 4096
#define NH 4
#define ND 32
#define NO 384
#define ATTN_SCALE 0.17677669529663687f  // 32^-0.5

#if defined(__AMDGCN__) && __has_builtin(__builtin_amdgcn_tensor_load_to_lds)
#define HAS_TDM 1
#else
#define HAS_TDM 0
#endif

static __device__ __forceinline__ unsigned short f2bf(float f) {
    unsigned u = __float_as_uint(f);
    unsigned r = u + 0x7FFFu + ((u >> 16) & 1u);   // round-to-nearest-even
    return (unsigned short)(r >> 16);
}

static __device__ __forceinline__ v16bf cat8(v8bf lo, v8bf hi) {
    return __builtin_shufflevector(lo, hi, 0,1,2,3,4,5,6,7,8,9,10,11,12,13,14,15);
}

static __device__ __forceinline__ v8f wmma_bf16(v16bf a, v16bf b, v8f c) {
    // v_wmma_f32_16x16x32_bf16: (neg_a, A, neg_b, B, c_mod, C, reuse_a, reuse_b)
    return __builtin_amdgcn_wmma_f32_16x16x32_bf16(false, a, false, b, (short)0, c, false, false);
}

#if HAS_TDM
// Issue one 2D TDM tile load: global (bf16 elements, row stride `stride_el`)
// -> LDS (packed rows). D# layout per CDNA5 ISA ch.8 (group0 + group1, groups 2/3 zero).
static __device__ __forceinline__ void tdm_load_2d(unsigned lds_off, const void* gptr,
                                                   unsigned tile_d0, unsigned tile_d1,
                                                   unsigned long long stride_el,
                                                   unsigned tensor_d0, unsigned tensor_d1) {
    unsigned long long ga = (unsigned long long)gptr;
    v4u g0;
    g0[0] = 1u;                                   // count=1, user descriptor
    g0[1] = lds_off;                              // lds_addr (bytes)
    g0[2] = (unsigned)ga;                         // global_addr[31:0]
    g0[3] = (unsigned)(ga >> 32) | (2u << 30);    // global_addr[56:32] | type=2
    v8i g1;
    g1[0] = (int)(1u << 16);                      // workgroup_mask=0, data_size=1 (2B)
    g1[1] = (int)((tensor_d0 & 0xFFFFu) << 16);   // tensor_dim0[15:0] @ bits 63:48
    g1[2] = (int)((tensor_d0 >> 16) | ((tensor_d1 & 0xFFFFu) << 16));  // dim0 hi | dim1 lo
    g1[3] = (int)((tensor_d1 >> 16) | (tile_d0 << 16));                // dim1 hi | tile_dim0
    g1[4] = (int)(tile_d1 & 0xFFFFu);             // tile_dim1 (tile_dim2=0)
    g1[5] = (int)(unsigned)(stride_el & 0xFFFFFFFFull);        // tensor_dim0_stride[31:0]
    g1[6] = (int)(unsigned)((stride_el >> 32) & 0xFFFFull);    // stride[47:32]
    g1[7] = 0;
    v4i z4 = {0, 0, 0, 0};
#if __clang_major__ <= 22
    __builtin_amdgcn_tensor_load_to_lds(g0, g1, z4, z4, 0);
#else
    v8i z8 = {0, 0, 0, 0, 0, 0, 0, 0};
    __builtin_amdgcn_tensor_load_to_lds(g0, g1, z4, z4, z8, 0);
#endif
}
#endif

// ---------------- 1) x[b][c][n] f32 -> Xt[b][n][c] bf16 ----------------
__global__ __launch_bounds__(256) void convert_x_kernel(const float* __restrict__ x,
                                                        unsigned short* __restrict__ Xt) {
    size_t idx = (size_t)blockIdx.x * 256 + threadIdx.x;   // < NB*NC*NN
    int n = (int)(idx % NN);
    size_t t = idx / NN;
    int c = (int)(t % NC);
    int b = (int)(t / NC);
    Xt[((size_t)b * NN + n) * NC + c] = f2bf(x[idx]);
}

// ---------------- 2) weights f32 -> bf16 (+ zero-bias scratch) ----------------
__global__ __launch_bounds__(256) void convert_w_kernel(const float* __restrict__ wqkv,
                                                        const float* __restrict__ wout,
                                                        unsigned short* __restrict__ wqkv_b,
                                                        unsigned short* __restrict__ wout_b,
                                                        float* __restrict__ zbias) {
    int idx = blockIdx.x * 256 + threadIdx.x;
    if (idx < NO) zbias[idx] = 0.f;
    if (idx < NO * NC) {
        wqkv_b[idx] = f2bf(wqkv[idx]);
    } else {
        int j = idx - NO * NC;          // < NC*NC
        wout_b[j] = f2bf(wout[j]);
    }
}

// ---------------- 3/6) GEMM: out[b][o][n] = sum_c W[o][c] * X[b][n][c] + bias ----------------
// One 16x16 output tile per wave; K accumulated in chunks of 32 with bf16 WMMA.
__global__ __launch_bounds__(256) void gemm_wmma_bf16_kernel(const unsigned short* __restrict__ Wb,
                                                             const unsigned short* __restrict__ Xb,
                                                             const float* __restrict__ bias,
                                                             float* __restrict__ out,
                                                             int Odim, int Cdim, int Nn) {
    const int lane = threadIdx.x & 31;
    const int wv   = threadIdx.x >> 5;
    const int hi   = lane >> 4;
    const int lr   = lane & 15;
    const int bIdx = blockIdx.z;
    const int o0   = blockIdx.y * 16;
    const int n0   = (blockIdx.x * 8 + wv) * 16;

    // A-matrix lane layout (16-bit 16x32): row M = lane&15; K range hi*16..+16 contiguous
    const unsigned short* wrow = Wb + (size_t)(o0 + lr) * Cdim + hi * 16;
    // B-matrix lane layout (16-bit 32x16): col N = lane&15; K range hi*16..+16 contiguous
    const unsigned short* xrow = Xb + ((size_t)bIdx * Nn + (n0 + lr)) * Cdim + hi * 16;

    v8f acc = {};
#pragma unroll
    for (int cc = 0; cc < 128; cc += 32) {
        v8bf alo = *(const v8bf*)(wrow + cc);
        v8bf ahi = *(const v8bf*)(wrow + cc + 8);
        v16bf a  = cat8(alo, ahi);
        v16bf b  = *(const v16bf*)(xrow + cc);
        acc = wmma_bf16(a, b, acc);
    }
    (void)Cdim;

#pragma unroll
    for (int e = 0; e < 8; ++e) {
        int M = e + 8 * hi;
        float r = acc[e] + bias[o0 + M];   // bias pointer always valid (zeros if none)
        out[((size_t)bIdx * Odim + (o0 + M)) * Nn + (n0 + lr)] = r;
    }
}

// ---------------- 4) marshal qkv f32 -> Q/K/V bf16 layouts ----------------
// Q[b][h][n][d] (scaled), K[b][h][n][d], V[b][h][d][n]
__global__ __launch_bounds__(256) void marshal_qkv_kernel(const float* __restrict__ qkv,
                                                          unsigned short* __restrict__ Q,
                                                          unsigned short* __restrict__ K,
                                                          unsigned short* __restrict__ V) {
    size_t idx = (size_t)blockIdx.x * 256 + threadIdx.x;   // < NB*NO*NN
    int n = (int)(idx % NN);
    size_t t = idx / NN;
    int o = (int)(t % NO);
    int b = (int)(t / NO);
    float v = qkv[idx];
    if (o < 128) {
        int h = o >> 5, d = o & 31;
        Q[(((size_t)b * NH + h) * NN + n) * ND + d] = f2bf(v * ATTN_SCALE);
    } else if (o < 256) {
        int oo = o - 128;
        int h = oo >> 5, d = oo & 31;
        K[(((size_t)b * NH + h) * NN + n) * ND + d] = f2bf(v);
    } else {
        int oo = o - 256;
        int h = oo >> 5, d = oo & 31;
        V[(((size_t)b * NH + h) * ND + d) * NN + n] = f2bf(v);
    }
}

// ---------------- 5) flash attention ----------------
// grid (NN/128, NH, NB), 256 threads = 8 waves; each wave owns a 16-row Q tile.
// K/V blocks staged into double-buffered LDS by the Tensor Data Mover (wave 0),
// overlapped with WMMA compute on the other buffer.
__global__ __launch_bounds__(256) void flash_attn_bf16_kernel(const unsigned short* __restrict__ Qg,
                                                              const unsigned short* __restrict__ Kg,
                                                              const unsigned short* __restrict__ Vg,
                                                              unsigned short* __restrict__ Og) {
    __shared__ unsigned short Kl[2][64 * 32];   // [buf][j][d]
    __shared__ unsigned short Vl[2][32 * 64];   // [buf][d][j]
    __shared__ unsigned short Pl[8][16 * 32];   // per-wave P staging [row][j]

    const int tid  = threadIdx.x;
    const int lane = tid & 31;
    const int wv   = tid >> 5;
    const int hi   = lane >> 4;
    const int lr   = lane & 15;
    const int b    = blockIdx.z;
    const int h    = blockIdx.y;
    const int qn   = blockIdx.x * 128 + wv * 16;
    const size_t bh = (size_t)b * NH + h;

    const unsigned short* Kbase = Kg + bh * (size_t)NN * ND;
    const unsigned short* Vbase = Vg + bh * (size_t)ND * NN;

    // Q A-tile (16x32, K=d): resident for the whole kernel
    const unsigned short* qrow = Qg + (bh * NN + qn + lr) * ND + hi * 16;
    v16bf aQ = cat8(*(const v8bf*)(qrow), *(const v8bf*)(qrow + 8));

    v8f acc0 = {};   // O columns d = 0..15
    v8f acc1 = {};   // O columns d = 16..31
    float m[8], l[8];
#pragma unroll
    for (int e = 0; e < 8; ++e) { m[e] = -1e30f; l[e] = 0.f; }

#if HAS_TDM
    if (wv == 0) {
        // prologue: DMA block 0 into buffer 0
        tdm_load_2d((unsigned)(unsigned long long)(uintptr_t)&Kl[0][0], Kbase,
                    /*tile*/ 2048, 1, /*stride*/ 2048, /*tensor*/ 2048, 1);
        tdm_load_2d((unsigned)(unsigned long long)(uintptr_t)&Vl[0][0], Vbase,
                    /*tile*/ 64, 32, /*stride*/ NN, /*tensor*/ 64, 32);
    }
#endif

    for (int kb = 0; kb < NN; kb += 64) {
        const int buf = (kb >> 6) & 1;
#if HAS_TDM
        if (wv == 0) {
            if (kb + 64 < NN) {
                // prefetch next block into the other buffer (safe: all waves passed
                // the trailing barrier of the previous iteration, so buf^1 is free)
                tdm_load_2d((unsigned)(unsigned long long)(uintptr_t)&Kl[buf ^ 1][0],
                            Kbase + (size_t)(kb + 64) * ND, 2048, 1, 2048, 2048, 1);
                tdm_load_2d((unsigned)(unsigned long long)(uintptr_t)&Vl[buf ^ 1][0],
                            Vbase + (kb + 64), 64, 32, NN, 64, 32);
                __builtin_amdgcn_s_wait_tensorcnt(2);   // current buffer's 2 DMAs done
            } else {
                __builtin_amdgcn_s_wait_tensorcnt(0);
            }
        }
#else
        // fallback: cooperative synchronous staging (16B per thread per buffer)
        ((ulonglong2*)Kl[buf])[tid] = ((const ulonglong2*)(Kbase + (size_t)kb * ND))[tid];
        {
            int d = tid >> 3, ch = tid & 7;
            ((ulonglong2*)Vl[buf])[tid] =
                *(const ulonglong2*)(Vbase + (size_t)d * NN + kb + ch * 8);
        }
#endif
        __syncthreads();

        const unsigned short* Kb = Kl[buf];
        const unsigned short* Vb = Vl[buf];

#pragma unroll
        for (int jc = 0; jc < 64; jc += 32) {
            // S = Q * K^T  (two 16x16 tiles covering 32 keys)
            v16bf bk0 = *(const v16bf*)&Kb[(jc + lr) * 32 + hi * 16];
            v16bf bk1 = *(const v16bf*)&Kb[(jc + 16 + lr) * 32 + hi * 16];
            v8f z = {};
            v8f s0 = wmma_bf16(aQ, bk0, z);
            v8f s1 = wmma_bf16(aQ, bk1, z);

            // online softmax: rows live across 16 lanes of one half-wave
            float bm[8];
#pragma unroll
            for (int e = 0; e < 8; ++e) bm[e] = fmaxf(s0[e], s1[e]);
#pragma unroll
            for (int off = 1; off <= 8; off <<= 1)
#pragma unroll
                for (int e = 0; e < 8; ++e) bm[e] = fmaxf(bm[e], __shfl_xor(bm[e], off, 32));

            float p0[8], p1[8], rs[8], al[8];
#pragma unroll
            for (int e = 0; e < 8; ++e) {
                float mn = fmaxf(m[e], bm[e]);
                al[e] = __expf(m[e] - mn);
                p0[e] = __expf(s0[e] - mn);
                p1[e] = __expf(s1[e] - mn);
                rs[e] = p0[e] + p1[e];
                m[e]  = mn;
            }
#pragma unroll
            for (int off = 1; off <= 8; off <<= 1)
#pragma unroll
                for (int e = 0; e < 8; ++e) rs[e] += __shfl_xor(rs[e], off, 32);
#pragma unroll
            for (int e = 0; e < 8; ++e) {
                l[e] = l[e] * al[e] + rs[e];
                acc0[e] *= al[e];
                acc1[e] *= al[e];
            }

            // restage P through per-wave LDS to A-matrix layout
            unsigned short* Ps = Pl[wv];
#pragma unroll
            for (int e = 0; e < 8; ++e) {
                int M = e + 8 * hi;
                Ps[M * 32 + lr]      = f2bf(p0[e]);
                Ps[M * 32 + 16 + lr] = f2bf(p1[e]);
            }
            asm volatile("s_wait_dscnt 0" ::: "memory");
            v16bf aP = cat8(*(const v8bf*)&Ps[lr * 32 + hi * 16],
                            *(const v8bf*)&Ps[lr * 32 + hi * 16 + 8]);

            // O += P * V  (two d halves)
            v16bf bv0 = *(const v16bf*)&Vb[lr * 64        + jc + hi * 16];
            v16bf bv1 = *(const v16bf*)&Vb[(16 + lr) * 64 + jc + hi * 16];
            acc0 = wmma_bf16(aP, bv0, acc0);
            acc1 = wmma_bf16(aP, bv1, acc1);
        }
        __syncthreads();
    }

    // normalize and emit O[b][n][h*32 + d] bf16 (c-contiguous for the out-proj GEMM)
#pragma unroll
    for (int e = 0; e < 8; ++e) {
        int M = e + 8 * hi;
        float inv = 1.f / l[e];
        unsigned short* orow = Og + ((size_t)b * NN + qn + M) * NC + h * ND;
        orow[lr]      = f2bf(acc0[e] * inv);
        orow[16 + lr] = f2bf(acc1[e] * inv);
    }
}

// ---------------- launch ----------------
extern "C" void kernel_launch(void* const* d_in, const int* in_sizes, int n_in,
                              void* d_out, int out_size, void* d_ws, size_t ws_size,
                              hipStream_t stream) {
    const float* x     = (const float*)d_in[0];   // [4,128,64,64]
    const float* wqkv  = (const float*)d_in[1];   // [384,128]
    const float* wout  = (const float*)d_in[2];   // [128,128]
    const float* bout  = (const float*)d_in[3];   // [128]
    float* out = (float*)d_out;                   // [4,128,64,64]

    char* ws = (char*)d_ws;
    size_t off = 0;
    unsigned short* Xt     = (unsigned short*)(ws + off); off += (size_t)NB * NN * NC * 2;  // 4 MB
    unsigned short* Wqkv_b = (unsigned short*)(ws + off); off += (size_t)NO * NC * 2;       // 96 KB
    unsigned short* Wout_b = (unsigned short*)(ws + off); off += (size_t)NC * NC * 2;       // 32 KB
    float*          zbias  = (float*)(ws + off);          off += 2048;                      // zero bias
    float*          qkv    = (float*)(ws + off);          off += (size_t)NB * NO * NN * 4;  // 25 MB
    unsigned short* Q      = (unsigned short*)(ws + off); off += (size_t)NB * NH * NN * ND * 2;
    unsigned short* K      = (unsigned short*)(ws + off); off += (size_t)NB * NH * NN * ND * 2;
    unsigned short* V      = (unsigned short*)(ws + off); off += (size_t)NB * NH * NN * ND * 2;
    unsigned short* O      = (unsigned short*)(ws + off); off += (size_t)NB * NN * NC * 2;

    // 1) transpose+convert x
    convert_x_kernel<<<(NB * NC * NN) / 256, 256, 0, stream>>>(x, Xt);
    // 2) convert weights (+ zero bias scratch)
    convert_w_kernel<<<(NO * NC + NC * NC) / 256, 256, 0, stream>>>(wqkv, wout, Wqkv_b, Wout_b, zbias);
    // 3) QKV projection: [384x128] x [128xN] per batch
    gemm_wmma_bf16_kernel<<<dim3(NN / 128, NO / 16, NB), 256, 0, stream>>>(
        Wqkv_b, Xt, zbias, qkv, NO, NC, NN);
    // 4) marshal Q/K/V
    marshal_qkv_kernel<<<(size_t)(NB * NO * NN) / 256, 256, 0, stream>>>(qkv, Q, K, V);
    // 5) flash attention (TDM double-buffered K/V staging)
    flash_attn_bf16_kernel<<<dim3(NN / 128, NH, NB), 256, 0, stream>>>(Q, K, V, O);
    // 6) output projection + bias
    gemm_wmma_bf16_kernel<<<dim3(NN / 128, NC / 16, NB), 256, 0, stream>>>(
        Wout_b, O, bout, out, NC, NC, NN);
}
